// ResidualAttentionBlock_adapter_601295421804
// MI455X (gfx1250) — compile-verified, matrix operand order checked by reading, not run on hardware
//
#include <hip/hip_runtime.h>
#include <hip/hip_bf16.h>
#include <math.h>
#include <stdint.h>

// ---------------------------------------------------------------------------
// Types
// ---------------------------------------------------------------------------
typedef _Float16 half_t;
typedef __attribute__((ext_vector_type(16))) _Float16 v16h;
typedef __attribute__((ext_vector_type(8)))  _Float16 v8h;
typedef __attribute__((ext_vector_type(8)))  float    v8f;
typedef __attribute__((ext_vector_type(4)))  unsigned v4u;
typedef __attribute__((ext_vector_type(8)))  unsigned v8u;

#define WMMA_F32_F16(a, b, c) \
  __builtin_amdgcn_wmma_f32_16x16x32_f16(false, (a), false, (b), (short)0, (c), false, false)

// Problem constants
#define D_MODEL 1024
#define NHEAD   16
#define HDIM    64
#define CA      384
#define TT      16
#define BV      4
#define LQ      257
#define BT      64            // BV*TT
#define MTOK    (BT * LQ)     // 16448
#define MPATCH  (BT * (LQ-1)) // 16384

// ---------------------------------------------------------------------------
// Fragment helpers (16-bit A/B matrix layout, wave32):
// lane holds row r = lane&15; k-halves k0 = (lane>>4)*8: [k0..k0+7] and [k0+16..k0+23]
// ---------------------------------------------------------------------------
__device__ __forceinline__ v16h mk16(v8h lo, v8h hi) {
  v16h r;
#pragma unroll
  for (int i = 0; i < 8; ++i) { r[i] = lo[i]; r[i + 8] = hi[i]; }
  return r;
}

__device__ __forceinline__ v16h frag_ld(const half_t* base, int ld, int lane) {
  const half_t* p = base + (size_t)(lane & 15) * ld + ((lane >> 4) << 3);
  return mk16(*(const v8h*)p, *(const v8h*)(p + 16));
}

__device__ __forceinline__ v16h frag_ld_clamp(const half_t* base, size_t ld, int r0,
                                              int lane, int maxRow) {
  int r = r0 + (lane & 15);
  if (r > maxRow) r = maxRow;
  const half_t* p = base + (size_t)r * ld + ((lane >> 4) << 3);
  return mk16(*(const v8h*)p, *(const v8h*)(p + 16));
}

// Async global->LDS 128-bit copy (CDNA5 ASYNC path, tracked by ASYNCcnt).
__device__ __forceinline__ void async_copy_b128(const half_t* gaddr, half_t* laddr) {
  unsigned l = (unsigned)(uintptr_t)laddr;  // low 32 bits of flat addr == LDS offset
  asm volatile("global_load_async_to_lds_b128 %0, %1, off"
               :: "v"(l), "v"(gaddr) : "memory");
}
__device__ __forceinline__ void wait_async_le2() {
  asm volatile("s_wait_asynccnt 0x2" ::: "memory");
}
__device__ __forceinline__ void wait_async_0() {
  asm volatile("s_wait_asynccnt 0x0" ::: "memory");
}

// ---------------------------------------------------------------------------
// Tensor Data Mover: 2D tile load global->LDS (TENSORcnt-tracked).
// D# per CDNA5 ISA. 2D tensor -> groups 2/3 NULL.
// Pads LDS dest every 16 DWORDs (64B row) by 4 DWORDs (16B) -> stride 80B.
// ---------------------------------------------------------------------------
__device__ __forceinline__ void tdm_load_2d(const half_t* gsrc, half_t* ldst,
                                            unsigned tileRows, unsigned tileCols,
                                            unsigned strideElems) {
  unsigned lds = (unsigned)(uintptr_t)ldst;
  unsigned long long ga = (unsigned long long)(uintptr_t)gsrc;
  v4u g0;
  g0[0] = 1u;                                                  // count=1 (valid), user mode
  g0[1] = lds;                                                 // lds_addr (bytes)
  g0[2] = (unsigned)ga;                                        // global_addr[31:0]
  g0[3] = (unsigned)((ga >> 32) & 0x01FFFFFFu) | (2u << 30);   // addr[56:32], type=2

  unsigned long long td0 = strideElems;       // tensor_dim0 (elements)
  unsigned long long td1 = 0x7FFFFFull;       // tensor_dim1: ample rows
  v8u g1;
  g1[0] = (1u << 16)        // data_size = 1 -> 2 bytes
        | (1u << 20)        // pad_enable
        | (3u << 22)        // pad_interval: 16 DWORDs
        | (3u << 25);       // pad_amount  : 4 DWORDs
  g1[1] = (unsigned)(td0 << 16);                               // [63:48] dim0 lo16
  g1[2] = (unsigned)((td0 >> 16) & 0xFFFFu) | (unsigned)(td1 << 16);
  g1[3] = (unsigned)((td1 >> 16) & 0xFFFFu) | (tileCols << 16);// [127:112] tile_dim0
  g1[4] = tileRows;                                            // [143:128] tile_dim1
  g1[5] = strideElems;                                         // dim0_stride lo32
  g1[6] = 0u;
  g1[7] = 0u;
  asm volatile("tensor_load_to_lds %0, %1" :: "s"(g0), "s"(g1) : "memory");
}

// ---------------------------------------------------------------------------
// fp32 -> f16 convert (weights)
// ---------------------------------------------------------------------------
__global__ void cvt_f16_kernel(const float* __restrict__ in, half_t* __restrict__ out, int n) {
  int i = blockIdx.x * 256 + threadIdx.x;
  if (i < n) out[i] = (half_t)in[i];
}

// ---------------------------------------------------------------------------
// LayerNorm (fp32 in, f16 out), one block per row, D=1024
// ---------------------------------------------------------------------------
__global__ __launch_bounds__(256) void ln_kernel(const float* __restrict__ x,
                                                 const float* __restrict__ g,
                                                 const float* __restrict__ b,
                                                 half_t* __restrict__ out, int D) {
  int row = blockIdx.x;
  const float* xr = x + (size_t)row * D;
  float s = 0.f, s2 = 0.f;
  for (int c = threadIdx.x; c < D; c += 256) { float v = xr[c]; s += v; s2 += v * v; }
#pragma unroll
  for (int o = 16; o; o >>= 1) { s += __shfl_down(s, o, 32); s2 += __shfl_down(s2, o, 32); }
  __shared__ float red0[8], red1[8];
  int wid = threadIdx.x >> 5;
  if ((threadIdx.x & 31) == 0) { red0[wid] = s; red1[wid] = s2; }
  __syncthreads();
  if (threadIdx.x == 0) {
    float a = 0.f, c2 = 0.f;
#pragma unroll
    for (int i = 0; i < 8; ++i) { a += red0[i]; c2 += red1[i]; }
    red0[0] = a; red1[0] = c2;
  }
  __syncthreads();
  float mu = red0[0] / D;
  float var = red1[0] / D - mu * mu;
  float rs = rsqrtf(var + 1e-5f);
  half_t* orow = out + (size_t)row * D;
  for (int c = threadIdx.x; c < D; c += 256)
    orow[c] = (half_t)((xr[c] - mu) * rs * g[c] + b[c]);
}

// ---------------------------------------------------------------------------
// WMMA GEMM: C[M,N] = A[M,K](f16 rowmajor) x W[N,K](f16 rowmajor)^T + bias
// block 256 thr = 8 waves; block tile 128x128; wave tile 32x64; K step 32.
// Full blocks: A via async-LDS DMA (all waves), W via TDM (wave 0), double-buffered.
// Edge block (M remainder): guarded synchronous staging.
// ---------------------------------------------------------------------------
#define BM 128
#define BN 128
#define BK 32
#define LDS_STRIDE 40  // halves (=80B): 64B row + 16B TDM pad

__global__ __launch_bounds__(256)
void gemm_wmma_kernel(const half_t* __restrict__ A, const half_t* __restrict__ W,
                      const float* __restrict__ bias, const float* __restrict__ resid,
                      float* __restrict__ outF, half_t* __restrict__ outH,
                      int M, int N, int K, int gelu) {
  __shared__ __align__(16) half_t As[2][BM * LDS_STRIDE];
  __shared__ __align__(16) half_t Ws[2][BN * LDS_STRIDE];

  const int tid = threadIdx.x, lane = tid & 31, w = tid >> 5;
  const int wm = w & 3;   // 4 waves along M (32 rows each)
  const int wn = w >> 2;  // 2 waves along N (64 cols each)
  const int bm = blockIdx.x * BM, bn = blockIdx.y * BN;

  v8f acc[2][4];
#pragma unroll
  for (int i = 0; i < 2; ++i)
#pragma unroll
    for (int j = 0; j < 4; ++j) acc[i][j] = (v8f){};

  auto compute_stage = [&](const half_t* Ab, const half_t* Wb) {
    v16h af[2], bf[4];
#pragma unroll
    for (int i = 0; i < 2; ++i)
      af[i] = frag_ld(Ab + (wm * 32 + i * 16) * LDS_STRIDE, LDS_STRIDE, lane);
#pragma unroll
    for (int j = 0; j < 4; ++j)
      bf[j] = frag_ld(Wb + (wn * 64 + j * 16) * LDS_STRIDE, LDS_STRIDE, lane);
#pragma unroll
    for (int i = 0; i < 2; ++i)
#pragma unroll
      for (int j = 0; j < 4; ++j)
        acc[i][j] = WMMA_F32_F16(af[i], bf[j], acc[i][j]);
  };

  const bool fullM = (bm + BM) <= M;
  if (fullM) {
    auto stage_A_async = [&](int buf, int k0) {  // 2 async ops / wave / stage
#pragma unroll
      for (int e = tid; e < (BM * BK) / 8; e += 256) {
        int r = e >> 2, grp = e & 3;
        async_copy_b128(A + (size_t)(bm + r) * K + k0 + grp * 8,
                        &As[buf][r * LDS_STRIDE + grp * 8]);
      }
    };
    auto stage_W_tdm = [&](int buf, int k0) {    // 1 TDM / stage, wave 0 only
      if (w == 0)
        tdm_load_2d(W + (size_t)bn * K + k0, &Ws[buf][0], BN, BK, (unsigned)K);
    };

    const int nstage = K / BK;
    stage_W_tdm(0, 0);
    stage_A_async(0, 0);
    for (int s = 0; s < nstage; ++s) {
      const int buf = s & 1;
      if (s + 1 < nstage) {
        stage_W_tdm(buf ^ 1, (s + 1) * BK);
        stage_A_async(buf ^ 1, (s + 1) * BK);
        wait_async_le2();                              // stage s A-DMAs done
        if (w == 0) __builtin_amdgcn_s_wait_tensorcnt(1);  // stage s TDM done
      } else {
        wait_async_0();
        if (w == 0) __builtin_amdgcn_s_wait_tensorcnt(0);
      }
      __syncthreads();                   // all waves' data for buf landed
      compute_stage(As[buf], Ws[buf]);
      __syncthreads();                   // buf^1 free for next issue
    }
  } else {
    // ---- guarded synchronous staging for the M-edge block ----
    for (int k0 = 0; k0 < K; k0 += BK) {
#pragma unroll
      for (int e = tid; e < (BM * BK) / 8; e += 256) {
        int r = e >> 2, grp = e & 3;
        int gr = bm + r;
        v8h val = {};
        if (gr < M) val = *(const v8h*)(A + (size_t)gr * K + k0 + grp * 8);
        *(v8h*)(&As[0][r * LDS_STRIDE + grp * 8]) = val;
      }
#pragma unroll
      for (int e = tid; e < (BN * BK) / 8; e += 256) {
        int r = e >> 2, grp = e & 3;
        v8h val = *(const v8h*)(W + (size_t)(bn + r) * K + k0 + grp * 8);
        *(v8h*)(&Ws[0][r * LDS_STRIDE + grp * 8]) = val;
      }
      __syncthreads();
      compute_stage(As[0], Ws[0]);
      __syncthreads();
    }
  }

  // Epilogue. C layout: col = lane&15, row = g + 8*(lane>>4)
#pragma unroll
  for (int i = 0; i < 2; ++i) {
#pragma unroll
    for (int j = 0; j < 4; ++j) {
      int col = bn + wn * 64 + j * 16 + (lane & 15);
      float bv = bias ? bias[col] : 0.f;
#pragma unroll
      for (int g = 0; g < 8; ++g) {
        int row = bm + wm * 32 + i * 16 + g + ((lane >> 4) << 3);
        if (row < M) {
          float v = acc[i][j][g] + bv;
          if (gelu) v = v / (1.f + __expf(-1.702f * v));  // QuickGELU
          if (resid) v += resid[(size_t)row * N + col];
          size_t idx = (size_t)row * N + col;
          if (outF) outF[idx] = v;
          if (outH) outH[idx] = (half_t)v;
        }
      }
    }
  }
}

// ---------------------------------------------------------------------------
// Attention: one block per (head, bt). 8 waves; each wave owns q-tiles w,w+8,w+16.
// V^T staged in LDS (for the PV wmma); K/Q fragments read from global (L2-hot).
// S^T = K x Q^T per 16x16 tile -> online softmax -> O^T += V^T x P^T.
// ---------------------------------------------------------------------------
#define LPAD 288   // 257 padded to 9 chunks of 32
#define VT_S 296   // halves stride for V^T rows (16B aligned, conflict padding)

__global__ __launch_bounds__(256)
void attn_wmma_kernel(const half_t* __restrict__ qkv, half_t* __restrict__ ao) {
  __shared__ __align__(16) half_t Vt[HDIM * VT_S];  // 64 x 288 (+pad) halves

  const int h = blockIdx.x, bt = blockIdx.y;
  const int tid = threadIdx.x, lane = tid & 31, w = tid >> 5;
  const half_t* base = qkv + (size_t)bt * LQ * (3 * D_MODEL);
  const half_t* kbase = base + D_MODEL + h * HDIM;
  const half_t* vbase = base + 2 * D_MODEL + h * HDIM;
  const half_t* qbase = base + h * HDIM;

  // stage V^T: Vt[d][j] = V[j][d]
  for (int e = tid; e < LPAD * 8; e += 256) {
    int j = e >> 3, grp = e & 7;
    v8h val = {};
    if (j < LQ) val = *(const v8h*)(vbase + (size_t)j * (3 * D_MODEL) + grp * 8);
#pragma unroll
    for (int i = 0; i < 8; ++i) Vt[(grp * 8 + i) * VT_S + j] = val[i];
  }
  __syncthreads();

  const float hs = 0.125f;  // 64^-0.5

  for (int qt = w; qt < 17; qt += 8) {
    // Q fragments (B role, lane = query index i, contiguous over head dim)
    int iq = qt * 16 + (lane & 15);
    int iqc = iq > LQ - 1 ? LQ - 1 : iq;
    const half_t* qp = qbase + (size_t)iqc * (3 * D_MODEL) + ((lane >> 4) << 3);
    v16h qf0 = mk16(*(const v8h*)qp, *(const v8h*)(qp + 16));          // d 0..31
    v16h qf1 = mk16(*(const v8h*)(qp + 32), *(const v8h*)(qp + 48));   // d 32..63

    float mi = -1e30f, li = 0.f;
    v8f o0 = (v8f){}, o1 = (v8f){}, o2 = (v8f){}, o3 = (v8f){};

    for (int jc = 0; jc < LPAD; jc += 32) {
      // S^T tiles: A = K rows (lane = key j), B = Q^T
      v16h ka0 = frag_ld_clamp(kbase,      3 * D_MODEL, jc,      lane, LQ - 1);
      v16h ka1 = frag_ld_clamp(kbase + 32, 3 * D_MODEL, jc,      lane, LQ - 1);
      v16h kb0 = frag_ld_clamp(kbase,      3 * D_MODEL, jc + 16, lane, LQ - 1);
      v16h kb1 = frag_ld_clamp(kbase + 32, 3 * D_MODEL, jc + 16, lane, LQ - 1);
      v8f s0 = (v8f){}, s1 = (v8f){};
      s0 = WMMA_F32_F16(ka0, qf0, s0);
      s0 = WMMA_F32_F16(ka1, qf1, s0);
      s1 = WMMA_F32_F16(kb0, qf0, s1);
      s1 = WMMA_F32_F16(kb1, qf1, s1);

      // scale + mask + tile max. S^T layout: row j = g + 8*(lane>>4), col i = lane&15
      float tm = -1e30f;
      int jrow = ((lane >> 4) << 3);
#pragma unroll
      for (int g = 0; g < 8; ++g) {
        int j0 = jc + g + jrow;
        int j1 = jc + 16 + g + jrow;
        float a = s0[g] * hs; if (j0 >= LQ) a = -1e30f; s0[g] = a;
        float b = s1[g] * hs; if (j1 >= LQ) b = -1e30f; s1[g] = b;
        tm = fmaxf(tm, fmaxf(a, b));
      }
      tm = fmaxf(tm, __shfl_xor(tm, 16, 32));
      float mn = fmaxf(mi, tm);
      float f = __expf(mi - mn);
      float ps = 0.f;
      v16h pb;  // exact B-fragment layout for the PV wmma (k = j within chunk)
#pragma unroll
      for (int g = 0; g < 8; ++g) {
        float p0 = __expf(s0[g] - mn);
        float p1 = __expf(s1[g] - mn);
        ps += p0 + p1;
        pb[g] = (half_t)p0;
        pb[8 + g] = (half_t)p1;
      }
      ps += __shfl_xor(ps, 16, 32);
      li = li * f + ps;
      mi = mn;
#pragma unroll
      for (int g = 0; g < 8; ++g) { o0[g] *= f; o1[g] *= f; o2[g] *= f; o3[g] *= f; }

      // O^T += V^T x P^T  (A = V^T rows = head-dim, k-contiguous in LDS)
      v16h va0 = frag_ld(Vt + (0 * 16) * VT_S + jc, VT_S, lane);
      v16h va1 = frag_ld(Vt + (1 * 16) * VT_S + jc, VT_S, lane);
      v16h va2 = frag_ld(Vt + (2 * 16) * VT_S + jc, VT_S, lane);
      v16h va3 = frag_ld(Vt + (3 * 16) * VT_S + jc, VT_S, lane);
      o0 = WMMA_F32_F16(va0, pb, o0);
      o1 = WMMA_F32_F16(va1, pb, o1);
      o2 = WMMA_F32_F16(va2, pb, o2);
      o3 = WMMA_F32_F16(va3, pb, o3);
    }

    float inv = li > 0.f ? 1.f / li : 0.f;
    int icol = qt * 16 + (lane & 15);
    if (icol < LQ) {
      half_t* op = ao + ((size_t)bt * LQ + icol) * D_MODEL + h * HDIM;
      int dr = ((lane >> 4) << 3);
      v8h r0, r1, r2, r3;  // 8 consecutive d per tile -> vector stores
#pragma unroll
      for (int g = 0; g < 8; ++g) {
        r0[g] = (half_t)(o0[g] * inv);
        r1[g] = (half_t)(o1[g] * inv);
        r2[g] = (half_t)(o2[g] * inv);
        r3[g] = (half_t)(o3[g] * inv);
      }
      *(v8h*)(op + 0 * 16 + dr) = r0;
      *(v8h*)(op + 1 * 16 + dr) = r1;
      *(v8h*)(op + 2 * 16 + dr) = r2;
      *(v8h*)(op + 3 * 16 + dr) = r3;
    }
  }
}

// ---------------------------------------------------------------------------
// Temporal diff for offset adapter: td[bt*256+p, :] = xs - former (f16)
// ---------------------------------------------------------------------------
__global__ void tdiff_kernel(const float* __restrict__ y1, half_t* __restrict__ td, size_t n) {
  size_t idx = (size_t)blockIdx.x * 256 + threadIdx.x;
  if (idx >= n) return;
  int c = (int)(idx & 1023);
  size_t r = idx >> 10;
  int p = (int)(r % 256);
  int bt = (int)(r / 256);
  int t = bt & 15;
  size_t cur = ((size_t)bt * LQ + 1 + p) * D_MODEL + c;
  size_t prev = (t == 0) ? cur : cur - (size_t)LQ * D_MODEL;
  td[idx] = (half_t)(y1[cur] - y1[prev]);
}

// Depthwise 3x3 spatial conv (per frame), NHWC, zero pad. in f32 -> out f16
__global__ void conv2d_dw_kernel(const float* __restrict__ in, const float* __restrict__ wt,
                                 const float* __restrict__ bias, half_t* __restrict__ out,
                                 size_t n) {
  size_t idx = (size_t)blockIdx.x * 256 + threadIdx.x;
  if (idx >= n) return;
  int c = (int)(idx % CA);
  size_t r = idx / CA;
  int x = (int)(r % 16); r /= 16;
  int y = (int)(r % 16); r /= 16;
  int bt = (int)r;
  float acc = bias[c];
#pragma unroll
  for (int ky = 0; ky < 3; ++ky) {
#pragma unroll
    for (int kx = 0; kx < 3; ++kx) {
      int yy = y + ky - 1, xx = x + kx - 1;
      if (yy >= 0 && yy < 16 && xx >= 0 && xx < 16)
        acc += wt[(ky * 3 + kx) * CA + c] *
               in[(((size_t)bt * 16 + yy) * 16 + xx) * CA + c];
    }
  }
  out[idx] = (half_t)acc;
}

// Depthwise temporal conv (k=3, zero pad) over t at fixed (b,l,c). in f32 -> out f16
__global__ void conv1d_dw_kernel(const float* __restrict__ in, const float* __restrict__ wt,
                                 const float* __restrict__ bias, half_t* __restrict__ out,
                                 size_t n) {
  size_t idx = (size_t)blockIdx.x * 256 + threadIdx.x;
  if (idx >= n) return;
  int c = (int)(idx % CA);
  size_t r = idx / CA;
  int l = (int)(r % LQ);
  int bt = (int)(r / LQ);
  int t = bt & 15, b = bt >> 4;
  float acc = bias[c];
#pragma unroll
  for (int k = 0; k < 3; ++k) {
    int tt = t + k - 1;
    if (tt >= 0 && tt < TT)
      acc += wt[k * CA + c] * in[(((size_t)(b * TT + tt)) * LQ + l) * CA + c];
  }
  out[idx] = (half_t)acc;
}

// x2 = y1 + ya + scatter(off at tokens 1..256)
__global__ void combine_kernel(const float* __restrict__ y1, const float* __restrict__ ya,
                               const float* __restrict__ offo, float* __restrict__ x2,
                               size_t n) {
  size_t idx = (size_t)blockIdx.x * 256 + threadIdx.x;
  if (idx >= n) return;
  int c = (int)(idx & 1023);
  size_t r = idx >> 10;
  int l = (int)(r % LQ);
  int bt = (int)(r / LQ);
  float v = y1[idx] + ya[idx];
  if (l >= 1) v += offo[((size_t)bt * 256 + (l - 1)) * D_MODEL + c];
  x2[idx] = v;
}

// ---------------------------------------------------------------------------
// Host side
// ---------------------------------------------------------------------------
static inline size_t align256(size_t v) { return (v + 255) & ~(size_t)255; }

extern "C" void kernel_launch(void* const* d_in, const int* in_sizes, int n_in,
                              void* d_out, int out_size, void* d_ws, size_t ws_size,
                              hipStream_t stream) {
  (void)in_sizes; (void)n_in; (void)out_size; (void)ws_size;
  const float* x        = (const float*)d_in[0];
  const float* ln1_g    = (const float*)d_in[1];
  const float* ln1_b    = (const float*)d_in[2];
  const float* w_in     = (const float*)d_in[3];
  const float* b_in     = (const float*)d_in[4];
  const float* w_out    = (const float*)d_in[5];
  const float* b_out    = (const float*)d_in[6];
  const float* o_fc1_w  = (const float*)d_in[7];
  const float* o_fc1_b  = (const float*)d_in[8];
  const float* o_conv_w = (const float*)d_in[9];
  const float* o_conv_b = (const float*)d_in[10];
  const float* o_fc2_w  = (const float*)d_in[11];
  const float* o_fc2_b  = (const float*)d_in[12];
  const float* a_fc1_w  = (const float*)d_in[13];
  const float* a_fc1_b  = (const float*)d_in[14];
  const float* a_conv_w = (const float*)d_in[15];
  const float* a_conv_b = (const float*)d_in[16];
  const float* a_fc2_w  = (const float*)d_in[17];
  const float* a_fc2_b  = (const float*)d_in[18];
  const float* ln2_g    = (const float*)d_in[19];
  const float* ln2_b    = (const float*)d_in[20];
  const float* mlp_fc_w = (const float*)d_in[21];
  const float* mlp_fc_b = (const float*)d_in[22];
  const float* mlp_pj_w = (const float*)d_in[23];
  const float* mlp_pj_b = (const float*)d_in[24];

  char* ws = (char*)d_ws;
  size_t off = 0;
  auto take = [&](size_t bytes) { size_t r = off; off += align256(bytes); return r; };

  // f16 weights
  half_t* wInH  = (half_t*)(ws + take((size_t)3 * D_MODEL * D_MODEL * 2));
  half_t* wOutH = (half_t*)(ws + take((size_t)D_MODEL * D_MODEL * 2));
  half_t* oF1H  = (half_t*)(ws + take((size_t)CA * D_MODEL * 2));
  half_t* oF2H  = (half_t*)(ws + take((size_t)D_MODEL * CA * 2));
  half_t* aF1H  = (half_t*)(ws + take((size_t)CA * D_MODEL * 2));
  half_t* aF2H  = (half_t*)(ws + take((size_t)D_MODEL * CA * 2));
  half_t* mFcH  = (half_t*)(ws + take((size_t)4 * D_MODEL * D_MODEL * 2));
  half_t* mPjH  = (half_t*)(ws + take((size_t)4 * D_MODEL * D_MODEL * 2));

  // Region A (33.7MB): h1 -> attn_out -> oc -> ac -> h2 (sequential lifetimes)
  char* regA = ws + take((size_t)MTOK * D_MODEL * 2);
  half_t* h1 = (half_t*)regA;
  half_t* ao = (half_t*)regA;
  half_t* oc = (half_t*)regA;
  half_t* ac = (half_t*)regA;
  half_t* h2 = (half_t*)regA;

  // Region B (134.8MB): qkv -> {y1h, td} -> m1
  char* regB = ws + take((size_t)MTOK * 4 * D_MODEL * 2);
  half_t* qkv = (half_t*)regB;
  half_t* y1h = (half_t*)regB;
  half_t* td  = (half_t*)(regB + align256((size_t)MTOK * D_MODEL * 2));
  half_t* m1  = (half_t*)regB;

  float* y1   = (float*)(ws + take((size_t)MTOK * D_MODEL * 4));
  // Region C: o1 -> a1
  float* o1   = (float*)(ws + take((size_t)MTOK * CA * 4));
  float* a1   = o1;
  float* offo = (float*)(ws + take((size_t)MPATCH * D_MODEL * 4));
  float* ya   = (float*)(ws + take((size_t)MTOK * D_MODEL * 4));
  float* x2   = (float*)(ws + take((size_t)MTOK * D_MODEL * 4));

  auto cvt = [&](const float* src, half_t* dst, int n) {
    cvt_f16_kernel<<<(n + 255) / 256, 256, 0, stream>>>(src, dst, n);
  };
  cvt(w_in,     wInH,  3 * D_MODEL * D_MODEL);
  cvt(w_out,    wOutH, D_MODEL * D_MODEL);
  cvt(o_fc1_w,  oF1H,  CA * D_MODEL);
  cvt(o_fc2_w,  oF2H,  D_MODEL * CA);
  cvt(a_fc1_w,  aF1H,  CA * D_MODEL);
  cvt(a_fc2_w,  aF2H,  D_MODEL * CA);
  cvt(mlp_fc_w, mFcH,  4 * D_MODEL * D_MODEL);
  cvt(mlp_pj_w, mPjH,  4 * D_MODEL * D_MODEL);

  auto gemm = [&](const half_t* A, const half_t* W, const float* bias, const float* resid,
                  float* oF, half_t* oH, int M, int N, int K, int gelu) {
    dim3 grid((M + BM - 1) / BM, N / BN);
    gemm_wmma_kernel<<<grid, 256, 0, stream>>>(A, W, bias, resid, oF, oH, M, N, K, gelu);
  };

  // 1) LN1 -> h1 (f16)
  ln_kernel<<<MTOK, 256, 0, stream>>>(x, ln1_g, ln1_b, h1, D_MODEL);
  // 2) qkv = h1 @ w_in^T + b_in (f16)
  gemm(h1, wInH, b_in, nullptr, nullptr, qkv, MTOK, 3 * D_MODEL, D_MODEL, 0);
  // 3) attention -> ao (f16)
  attn_wmma_kernel<<<dim3(NHEAD, BT), 256, 0, stream>>>(qkv, ao);
  // 4) y1 = x + ao @ w_out^T + b_out (f32 + f16 copy)
  gemm(ao, wOutH, b_out, x, y1, y1h, MTOK, D_MODEL, D_MODEL, 0);
  // 5) offset adapter
  {
    size_t n = (size_t)MPATCH * D_MODEL;
    tdiff_kernel<<<(unsigned)((n + 255) / 256), 256, 0, stream>>>(y1, td, n);
  }
  gemm(td, oF1H, o_fc1_b, nullptr, o1, nullptr, MPATCH, CA, D_MODEL, 0);
  {
    size_t n = (size_t)MPATCH * CA;
    conv2d_dw_kernel<<<(unsigned)((n + 255) / 256), 256, 0, stream>>>(o1, o_conv_w, o_conv_b, oc, n);
  }
  gemm(oc, oF2H, o_fc2_b, nullptr, offo, nullptr, MPATCH, D_MODEL, CA, 0);
  // 6) temporal adapter
  gemm(y1h, aF1H, a_fc1_b, nullptr, a1, nullptr, MTOK, CA, D_MODEL, 0);
  {
    size_t n = (size_t)MTOK * CA;
    conv1d_dw_kernel<<<(unsigned)((n + 255) / 256), 256, 0, stream>>>(a1, a_conv_w, a_conv_b, ac, n);
  }
  gemm(ac, aF2H, a_fc2_b, nullptr, ya, nullptr, MTOK, D_MODEL, CA, 0);
  // 7) x2 = y1 + ya + scatter(offo)
  {
    size_t n = (size_t)MTOK * D_MODEL;
    combine_kernel<<<(unsigned)((n + 255) / 256), 256, 0, stream>>>(y1, ya, offo, x2, n);
  }
  // 8) MLP
  ln_kernel<<<MTOK, 256, 0, stream>>>(x2, ln2_g, ln2_b, h2, D_MODEL);
  gemm(h2, mFcH, mlp_fc_b, nullptr, nullptr, m1, MTOK, 4 * D_MODEL, D_MODEL, 1);
  gemm(m1, mPjH, mlp_pj_b, x2, (float*)d_out, nullptr, MTOK, D_MODEL, 4 * D_MODEL, 0);
}
